// L_ent_3642132267129
// MI455X (gfx1250) — compile-verified, matrix operand order checked by reading, not run on hardware
//
#include <hip/hip_runtime.h>

typedef __attribute__((ext_vector_type(16))) _Float16 v16h;
typedef __attribute__((ext_vector_type(8)))  float    v8f;

#define BINS        256
#define SIGMA_      30.0f
#define DELTA_      (1.0f / 256.0f)
#define LOG2E_      1.4426950408889634f
#define HW_         (512 * 512)
#define CHUNK       4096          // pixels staged per workgroup
#define BLOCK       256           // 8 waves (wave32)

__global__ void zero_hist_kernel(float* hist, int n) {
    int i = blockIdx.x * BLOCK + threadIdx.x;
    if (i < n) hist[i] = 0.0f;
}

// Each block: stage e = exp(-sigma*y) for CHUNK pixels into LDS, then each of
// the 8 waves reduces 2 bin-groups (16 bins each) over those pixels using
// v_wmma_f32_16x16x32_f16 with a ones A-matrix (D[m,n] = sum_k K[k,n]).
__global__ void __launch_bounds__(BLOCK)
soft_hist_kernel(const float* __restrict__ y, float* __restrict__ hist,
                 int chunks_per_batch) {
    __shared__ float sE[CHUNK];

    const int chunkId = blockIdx.x;
    const int batch   = chunkId / chunks_per_batch;
    const float* yp   = y + (size_t)chunkId * CHUNK;

    // ---- Phase 1: e = exp2(-sigma*log2e * y), 16 pixels/thread, float4 loads
    const float escale = -SIGMA_ * LOG2E_;
    const float4* y4 = (const float4*)yp;
    #pragma unroll
    for (int v = 0; v < CHUNK / (BLOCK * 4); ++v) {
        int idx = threadIdx.x + v * BLOCK;
        float4 d = y4[idx];
        sE[idx * 4 + 0] = __builtin_amdgcn_exp2f(escale * d.x);
        sE[idx * 4 + 1] = __builtin_amdgcn_exp2f(escale * d.y);
        sE[idx * 4 + 2] = __builtin_amdgcn_exp2f(escale * d.z);
        sE[idx * 4 + 3] = __builtin_amdgcn_exp2f(escale * d.w);
    }
    __syncthreads();

    // ---- Phase 2: WMMA reduction. wave handles bin-groups {wave, wave+8}.
    const int wave = threadIdx.x >> 5;       // 0..7
    const int lane = threadIdx.x & 31;       // wave32 lane
    const int half = lane >> 4;              // which 16-pixel half of the K=32 tile
    const int col  = lane & 15;              // bin within group (B-matrix column)

    v16h Aones;
    #pragma unroll
    for (int i = 0; i < 16; ++i) Aones[i] = (_Float16)1.0f;

    const float binscale = SIGMA_ * DELTA_ * LOG2E_;   // exp(sigma*delta*j) via exp2

    #pragma unroll
    for (int gi = 0; gi < 2; ++gi) {
        const int group = wave + gi * 8;               // 0..15
        const int bin   = group * 16 + col;
        // per-bin constants: A=exp(sd*j), B=exp(sd*(j+1))
        const float aj = __builtin_amdgcn_exp2f((float)bin       * binscale);
        const float bj = __builtin_amdgcn_exp2f((float)(bin + 1) * binscale);
        const float S = aj + bj;       // e-linear term
        const float P = aj * bj;       // e^2 term (max ~1.1e26, fits f32)
        const float W = bj - aj;       // numerator weight

        v8f acc = {};
        for (int t = 0; t < CHUNK / 32; ++t) {
            const int pbase = t * 32 + half * 16;
            v16h K;
            #pragma unroll
            for (int i = 0; i < 16; ++i) {
                // k = e*W / (1 + e*S + e^2*P): 2 fma + 2 mul + 1 v_rcp_f32
                float e   = sE[pbase + i];             // broadcast across half-wave
                float den = __builtin_fmaf(e, __builtin_fmaf(e, P, S), 1.0f);
                float k   = e * W * __builtin_amdgcn_rcpf(den);
                K[i] = (_Float16)k;
            }
            // D = ones(16x32) x K(32x16) + C  -> every row = per-bin column sums
            acc = __builtin_amdgcn_wmma_f32_16x16x32_f16(
                false, Aones, false, K, (short)0, acc, false, false);
        }
        // all 16x16 D entries in a column are identical copies of the bin sum;
        // lanes 0..15 (row m=0) commit one copy per bin.
        if (lane < 16) {
            atomicAdd(&hist[batch * BINS + bin], acc[0]);
        }
    }
}

__global__ void __launch_bounds__(BLOCK)
entropy_kernel(const float* __restrict__ hist, float* __restrict__ out, int nbins) {
    __shared__ float red[BLOCK];
    const float invN = 1.0f / (float)HW_;
    float local = 0.0f;
    for (int i = threadIdx.x; i < nbins; i += BLOCK) {
        float p  = hist[i] * invN + 1e-6f;
        float ln = __builtin_amdgcn_logf(p) * 0.6931471805599453f;  // log2 -> ln
        local   += -p * ln;
    }
    red[threadIdx.x] = local;
    __syncthreads();
    #pragma unroll
    for (int s = BLOCK / 2; s > 0; s >>= 1) {
        if (threadIdx.x < (unsigned)s) red[threadIdx.x] += red[threadIdx.x + s];
        __syncthreads();
    }
    if (threadIdx.x == 0) out[0] = 1.0f / red[0];
}

extern "C" void kernel_launch(void* const* d_in, const int* in_sizes, int n_in,
                              void* d_out, int out_size, void* d_ws, size_t ws_size,
                              hipStream_t stream) {
    const float* y = (const float*)d_in[0];
    float* out     = (float*)d_out;
    float* hist    = (float*)d_ws;             // first 4KB of scratch: (batches*256) f32

    const int total   = in_sizes[0];           // 4*1*512*512 = 1048576
    const int batches = total / HW_;           // 4
    const int nbinsT  = batches * BINS;        // 1024
    const int nchunks = total / CHUNK;         // 256
    const int cpb     = HW_ / CHUNK;           // 64 chunks per batch

    zero_hist_kernel<<<(nbinsT + BLOCK - 1) / BLOCK, BLOCK, 0, stream>>>(hist, nbinsT);
    soft_hist_kernel<<<nchunks, BLOCK, 0, stream>>>(y, hist, cpb);
    entropy_kernel<<<1, BLOCK, 0, stream>>>(hist, out, nbinsT);
}
